// IEGMNetXNOR_25864293057022
// MI455X (gfx1250) — 8-bit matrix operand order repaired and checked by reading, compiled, not run on hardware
//
#include <hip/hip_runtime.h>
#include <stdint.h>

typedef __attribute__((ext_vector_type(8))) int v8i;
typedef __attribute__((ext_vector_type(2))) int v2i;

#define EPS 1e-5f

constexpr int BATCH = 2048;
constexpr int H1S = 640;
constexpr int H2S = 336;
constexpr int H3S = 176;
constexpr int H4S = 96;
constexpr int H5S = 48;

constexpr size_t OFF_ACT1  = 0;
constexpr size_t SZ_ACT1   = (size_t)BATCH * H1S * 32;
constexpr size_t OFF_ACT2  = OFF_ACT1 + SZ_ACT1;
constexpr size_t SZ_ACT2   = (size_t)BATCH * H2S * 32;
constexpr size_t OFF_ACT3  = OFF_ACT2 + SZ_ACT2;
constexpr size_t SZ_ACT3   = (size_t)BATCH * H3S * 32;
constexpr size_t OFF_ACT4  = OFF_ACT3 + SZ_ACT3;
constexpr size_t SZ_ACT4   = (size_t)BATCH * H4S * 32;
constexpr size_t OFF_ACT5  = OFF_ACT4 + SZ_ACT4;
constexpr size_t SZ_ACT5   = (size_t)BATCH * H5S * 32;
constexpr size_t OFF_F1S   = OFF_ACT5 + SZ_ACT5;
constexpr size_t SZ_F1S    = (size_t)BATCH * 32;
constexpr size_t OFF_FRAGS = (OFF_F1S + SZ_F1S + 255) & ~(size_t)255;
constexpr size_t SZ_FRAGS  = 54 * 1024;
constexpr size_t OFF_THR   = (OFF_FRAGS + SZ_FRAGS + 255) & ~(size_t)255;
constexpr size_t SZ_THR    = 6 * 32 * sizeof(float);
constexpr size_t OFF_W1S   = (OFF_THR + SZ_THR + 255) & ~(size_t)255;
constexpr size_t SZ_W1S    = 32 * 6 * sizeof(float);
constexpr size_t OFF_W2S   = (OFF_W1S + SZ_W1S + 255) & ~(size_t)255;

constexpr int FRAG_L2 = 0, FRAG_L3 = 6, FRAG_L4 = 10, FRAG_L5 = 14, FRAG_FC1 = 18;

__global__ void prep_scalars(const float* __restrict__ bn_g, const float* __restrict__ bn_b,
                             const float* __restrict__ bn_m, const float* __restrict__ bn_v,
                             const float* __restrict__ g6, const float* __restrict__ b6,
                             const float* __restrict__ m6, const float* __restrict__ v6,
                             const float* __restrict__ w1, const float* __restrict__ fc2_w,
                             float* __restrict__ thr, float* __restrict__ w1s,
                             int8_t* __restrict__ w2s) {
  int t = threadIdx.x;
  if (t < 160) {
    thr[t] = bn_m[t] - bn_b[t] * sqrtf(bn_v[t] + EPS) / bn_g[t];
  } else if (t < 192) {
    int c = t - 160;
    thr[160 + c] = m6[c] - b6[c] * sqrtf(v6[c] + EPS) / g6[c];
  }
  if (t < 192) w1s[t] = (w1[t] >= 0.f) ? 1.f : -1.f;
  if (t < 64)  w2s[t] = (fc2_w[t] >= 0.f) ? (int8_t)1 : (int8_t)-1;
}

__global__ void prep_frags(const float* __restrict__ w2, const float* __restrict__ w3,
                           const float* __restrict__ w4, const float* __restrict__ w5,
                           const float* __restrict__ fc1_w, int8_t* __restrict__ frags) {
  int idx = blockIdx.x * blockDim.x + threadIdx.x;
  if (idx >= 54 * 32) return;
  int frag = idx >> 5, lane = idx & 31;
  const float* w; int kh; int chunk, ntile; bool is_fc = false;
  if (frag < 6)        { w = w2;    kh = 5; int f = frag;      chunk = f >> 1; ntile = f & 1; }
  else if (frag < 10)  { w = w3;    kh = 4; int f = frag - 6;  chunk = f >> 1; ntile = f & 1; }
  else if (frag < 14)  { w = w4;    kh = 4; int f = frag - 10; chunk = f >> 1; ntile = f & 1; }
  else if (frag < 18)  { w = w5;    kh = 4; int f = frag - 14; chunk = f >> 1; ntile = f & 1; }
  else                 { w = fc1_w; kh = 0; int f = frag - 18; chunk = f >> 1; ntile = f & 1; is_fc = true; }
  int hi = lane >> 4, n = ntile * 16 + (lane & 15);
  int8_t bytes[32];
  #pragma unroll
  for (int r = 0; r < 8; ++r)
    #pragma unroll
    for (int j = 0; j < 4; ++j) {
      int k = chunk * 64 + ((r >> 1) << 4) + (hi << 3) + ((r & 1) << 2) + j;
      int8_t v = 0;
      if (is_fc) {
        if (k < 1120) {
          int h = k >> 5, c = k & 31;
          v = (fc1_w[n * 1120 + c * 35 + h] >= 0.f) ? 1 : -1;
        }
      } else {
        if (k < kh * 32) {
          int c = k & 31, tap = k >> 5;
          v = (w[(n * 32 + c) * kh + tap] >= 0.f) ? 1 : -1;
        }
      }
      bytes[r * 4 + j] = v;
    }
  int8_t* dst = frags + (size_t)frag * 1024 + lane * 32;
  #pragma unroll
  for (int i = 0; i < 32; ++i) dst[i] = bytes[i];
}

__global__ void conv1_kernel(const float* __restrict__ x, const float* __restrict__ w1s,
                             const float* __restrict__ thr, int8_t* __restrict__ act1) {
  __shared__ float sw[192];
  __shared__ float st[32];
  int t = threadIdx.x;
  if (t < 192) sw[t] = w1s[t];
  if (t < 32)  st[t] = thr[t];
  __syncthreads();
  int hp = blockIdx.x * blockDim.x + t;
  int b  = blockIdx.y;
  if (hp >= 621) return;
  const float* xp = x + (long)b * 1248 + 2 * hp;
  float xs[7];
  #pragma unroll
  for (int j = 0; j < 7; ++j) xs[j] = xp[j];
  int words[8];
  #pragma unroll
  for (int g = 0; g < 8; ++g) {
    int wrd = 0;
    #pragma unroll
    for (int q = 0; q < 4; ++q) {
      int c = g * 4 + q;
      float y0 = 0.f, y1 = 0.f;
      #pragma unroll
      for (int j = 0; j < 6; ++j) { float w = sw[c * 6 + j]; y0 += w * xs[j]; y1 += w * xs[j + 1]; }
      float tc = st[c];
      int byte = ((y0 >= tc) || (y1 >= tc)) ? 1 : 0xFF;
      wrd |= (byte & 0xFF) << (8 * q);
    }
    words[g] = wrd;
  }
  int4* dst = (int4*)(act1 + ((long)b * H1S + hp) * 32);
  dst[0] = make_int4(words[0], words[1], words[2], words[3]);
  dst[1] = make_int4(words[4], words[5], words[6], words[7]);
}

template<int NCHUNKS, int KTOTAL, int HIN_S, int HPOOL, int HOUT_S, int NTILES>
__global__ void binconv_wmma(const int8_t* __restrict__ actIn, int8_t* __restrict__ actOut,
                             const int8_t* __restrict__ frags, const float* __restrict__ thr) {
  int lane = threadIdx.x & 31;
  int tile = blockIdx.x * 8 + (threadIdx.x >> 5);
  if (tile >= NTILES) return;
  int b = blockIdx.y;
  int h0 = tile * 16;
  int hi = lane >> 4, mrow = lane & 15;
  const int8_t* rowbase = actIn + ((long)b * HIN_S + h0 + mrow) * 32 + hi * 8;
  v8i c0 = {}, c1 = {};
  #pragma unroll
  for (int ch = 0; ch < NCHUNKS; ++ch) {
    v8i a;
    #pragma unroll
    for (int p = 0; p < 4; ++p) {
      int kg = ch * 64 + p * 16;
      v2i v = {};
      if (kg < KTOTAL) v = *(const v2i*)(rowbase + kg);
      a[2 * p]     = v.x;
      a[2 * p + 1] = v.y;
    }
    v8i b0 = *(const v8i*)(frags + (size_t)(ch * 2 + 0) * 1024 + lane * 32);
    v8i b1 = *(const v8i*)(frags + (size_t)(ch * 2 + 1) * 1024 + lane * 32);
    c0 = __builtin_amdgcn_wmma_i32_16x16x64_iu8(true, a, true, b0, c0, false, false);
    c1 = __builtin_amdgcn_wmma_i32_16x16x64_iu8(true, a, true, b1, c1, false, false);
  }
  int n = lane & 15;
  float t0 = thr[n], t1 = thr[16 + n];
  #pragma unroll
  for (int tq = 0; tq < 4; ++tq) {
    int hp = (h0 >> 1) + tq + hi * 4;
    if (hp < HPOOL) {
      int8_t o0 = (((float)c0[2 * tq] >= t0) || ((float)c0[2 * tq + 1] >= t0)) ? 1 : -1;
      int8_t o1 = (((float)c1[2 * tq] >= t1) || ((float)c1[2 * tq + 1] >= t1)) ? 1 : -1;
      int8_t* op = actOut + ((long)b * HOUT_S + hp) * 32;
      op[n] = o0; op[16 + n] = o1;
    }
  }
}

__global__ void fc1_wmma(const int8_t* __restrict__ act5, const int8_t* __restrict__ frags,
                         const float* __restrict__ t6, int8_t* __restrict__ f1s) {
  int wave = blockIdx.x * 8 + (threadIdx.x >> 5);
  int lane = threadIdx.x & 31;
  int hi = lane >> 4, mrow = lane & 15;
  const int8_t* rowbase = act5 + (long)(wave * 16 + mrow) * H5S * 32 + hi * 8;
  v8i c0 = {}, c1 = {};
  #pragma unroll
  for (int ch = 0; ch < 18; ++ch) {
    v8i a;
    #pragma unroll
    for (int p = 0; p < 4; ++p) {
      int kg = ch * 64 + p * 16;
      v2i v = {};
      if (kg < 1120) v = *(const v2i*)(rowbase + kg);
      a[2 * p]     = v.x;
      a[2 * p + 1] = v.y;
    }
    v8i b0 = *(const v8i*)(frags + (size_t)(ch * 2 + 0) * 1024 + lane * 32);
    v8i b1 = *(const v8i*)(frags + (size_t)(ch * 2 + 1) * 1024 + lane * 32);
    c0 = __builtin_amdgcn_wmma_i32_16x16x64_iu8(true, a, true, b0, c0, false, false);
    c1 = __builtin_amdgcn_wmma_i32_16x16x64_iu8(true, a, true, b1, c1, false, false);
  }
  int n = lane & 15;
  float th0 = t6[n], th1 = t6[16 + n];
  #pragma unroll
  for (int r = 0; r < 8; ++r) {
    int bm = wave * 16 + r + 8 * hi;
    f1s[bm * 32 + n]      = ((float)c0[r] >= th0) ? 1 : -1;
    f1s[bm * 32 + 16 + n] = ((float)c1[r] >= th1) ? 1 : -1;
  }
}

__global__ void fc2_kernel(const int8_t* __restrict__ f1s, const int8_t* __restrict__ w2s,
                           const float* __restrict__ g7, const float* __restrict__ b7,
                           const float* __restrict__ m7, const float* __restrict__ v7,
                           float* __restrict__ out) {
  int b = blockIdx.x * blockDim.x + threadIdx.x;
  if (b >= BATCH) return;
  const int8_t* f = f1s + (long)b * 32;
  #pragma unroll
  for (int o = 0; o < 2; ++o) {
    int acc = 0;
    #pragma unroll
    for (int k = 0; k < 32; ++k) acc += (int)f[k] * (int)w2s[o * 32 + k];
    out[b * 2 + o] = ((float)acc - m7[o]) * (g7[o] * rsqrtf(v7[o] + EPS)) + b7[o];
  }
}

extern "C" void kernel_launch(void* const* d_in, const int* in_sizes, int n_in,
                              void* d_out, int out_size, void* d_ws, size_t ws_size,
                              hipStream_t stream) {
  const float* x     = (const float*)d_in[0];
  const float* w1    = (const float*)d_in[1];
  const float* w2    = (const float*)d_in[2];
  const float* w3    = (const float*)d_in[3];
  const float* w4    = (const float*)d_in[4];
  const float* w5    = (const float*)d_in[5];
  const float* fc1_w = (const float*)d_in[6];
  const float* fc2_w = (const float*)d_in[7];
  const float* bn_g  = (const float*)d_in[8];
  const float* bn_b  = (const float*)d_in[9];
  const float* bn_m  = (const float*)d_in[10];
  const float* bn_v  = (const float*)d_in[11];
  const float* g6    = (const float*)d_in[12];
  const float* b6    = (const float*)d_in[13];
  const float* m6    = (const float*)d_in[14];
  const float* v6    = (const float*)d_in[15];
  const float* g7    = (const float*)d_in[16];
  const float* b7    = (const float*)d_in[17];
  const float* m7    = (const float*)d_in[18];
  const float* v7    = (const float*)d_in[19];
  float* out = (float*)d_out;

  uint8_t* ws   = (uint8_t*)d_ws;
  int8_t* act1  = (int8_t*)(ws + OFF_ACT1);
  int8_t* act2  = (int8_t*)(ws + OFF_ACT2);
  int8_t* act3  = (int8_t*)(ws + OFF_ACT3);
  int8_t* act4  = (int8_t*)(ws + OFF_ACT4);
  int8_t* act5  = (int8_t*)(ws + OFF_ACT5);
  int8_t* f1s   = (int8_t*)(ws + OFF_F1S);
  int8_t* frags = (int8_t*)(ws + OFF_FRAGS);
  float*  thr   = (float*)(ws + OFF_THR);
  float*  w1s   = (float*)(ws + OFF_W1S);
  int8_t* w2s   = (int8_t*)(ws + OFF_W2S);

  prep_scalars<<<1, 256, 0, stream>>>(bn_g, bn_b, bn_m, bn_v, g6, b6, m6, v6,
                                      w1, fc2_w, thr, w1s, w2s);
  prep_frags<<<(54 * 32 + 255) / 256, 256, 0, stream>>>(w2, w3, w4, w5, fc1_w, frags);

  conv1_kernel<<<dim3((621 + 255) / 256, BATCH), 256, 0, stream>>>(x, w1s, thr + 0 * 32, act1);

  binconv_wmma<3, 160, H1S, 308, H2S, 39><<<dim3(5, BATCH), 256, 0, stream>>>(
      act1, act2, frags + FRAG_L2 * 1024, thr + 1 * 32);
  binconv_wmma<2, 128, H2S, 152, H3S, 19><<<dim3(3, BATCH), 256, 0, stream>>>(
      act2, act3, frags + FRAG_L3 * 1024, thr + 2 * 32);
  binconv_wmma<2, 128, H3S, 74, H4S, 10><<<dim3(2, BATCH), 256, 0, stream>>>(
      act3, act4, frags + FRAG_L4 * 1024, thr + 3 * 32);
  binconv_wmma<2, 128, H4S, 35, H5S, 5><<<dim3(1, BATCH), 256, 0, stream>>>(
      act4, act5, frags + FRAG_L5 * 1024, thr + 4 * 32);
  fc1_wmma<<<(BATCH / 16) / 8, 256, 0, stream>>>(act5, frags + FRAG_FC1 * 1024, thr + 5 * 32, f1s);
  fc2_kernel<<<BATCH / 256, 256, 0, stream>>>(f1s, w2s, g7, b7, m7, v7, out);
}
